// Attention_5007931867377
// MI455X (gfx1250) — compile-verified, matrix operand order checked by reading, not run on hardware
//
#include <hip/hip_runtime.h>
#include <hip/hip_bf16.h>

// ---------------------------------------------------------------------------
// ViT attention forward on gfx1250 (MI455X), all matmuls via
// v_wmma_f32_16x16x32_bf16, flash-attention streaming softmax.
// B=8, N=1568, C=384, H=6, D=64.
// ---------------------------------------------------------------------------

typedef __bf16 bf16_t;
typedef __attribute__((ext_vector_type(16))) __bf16 v16bf;
typedef __attribute__((ext_vector_type(8)))  float  v8f;

#define WMMA_BF16(a, b, c) \
  __builtin_amdgcn_wmma_f32_16x16x32_bf16(false, (a), false, (b), (short)0, (c), false, false)

constexpr int B_  = 8;
constexpr int N_  = 1568;
constexpr int C_  = 384;
constexpr int H_  = 6;
constexpr int D_  = 64;
constexpr int BH  = B_ * H_;     // 48
constexpr int QT  = N_ / 16;     // 98 query tiles
constexpr int KGn = N_ / 32;     // 49 key groups of 32
constexpr float SCALE = 0.125f;  // D^-0.5
constexpr float EPS   = 1e-6f;
constexpr float EPSN  = 1e-6f / 1568.0f;

// Workspace layout (bytes). Fragments are [tile][kchunk][lane(32)][16 bf16].
constexpr size_t SZ_XF   = (size_t)784 * 12 * 32 * 16 * 2;     // 9,633,792
constexpr size_t SZ_WQKV = (size_t)72  * 12 * 32 * 16 * 2;     //   884,736
constexpr size_t SZ_WPRJ = (size_t)24  * 12 * 32 * 16 * 2;     //   294,912
constexpr size_t SZ_QF   = (size_t)BH * QT * 2 * 32 * 16 * 2;  // 9,633,792
constexpr size_t SZ_KF   = SZ_QF;
constexpr size_t SZ_VF   = (size_t)BH * KGn * 4 * 32 * 16 * 2; // 9,633,792
constexpr size_t SZ_QKV  = (size_t)12544 * 1152 * 4;           // 57,802,752
constexpr size_t SZ_O    = (size_t)12544 * 384 * 4;            // 19,267,584

constexpr size_t OFF_XF   = 0;
constexpr size_t OFF_WQKV = OFF_XF   + SZ_XF;
constexpr size_t OFF_WPRJ = OFF_WQKV + SZ_WQKV;
constexpr size_t OFF_QF   = OFF_WPRJ + SZ_WPRJ;
constexpr size_t OFF_KF   = OFF_QF   + SZ_QF;
constexpr size_t OFF_VF   = OFF_KF   + SZ_KF;
constexpr size_t OFF_QKV  = OFF_VF   + SZ_VF;
constexpr size_t OFF_O    = OFF_QKV;              // reuse dead QKV region
constexpr size_t OFF_OF   = OFF_O + SZ_O;         // still inside QKV region

// ---------------------------------------------------------------------------
// helpers
// ---------------------------------------------------------------------------
__device__ inline v16bf pack16(const float4& a, const float4& b,
                               const float4& c, const float4& d) {
  v16bf f;
  f[0]=(bf16_t)a.x; f[1]=(bf16_t)a.y; f[2]=(bf16_t)a.z; f[3]=(bf16_t)a.w;
  f[4]=(bf16_t)b.x; f[5]=(bf16_t)b.y; f[6]=(bf16_t)b.z; f[7]=(bf16_t)b.w;
  f[8]=(bf16_t)c.x; f[9]=(bf16_t)c.y; f[10]=(bf16_t)c.z; f[11]=(bf16_t)c.w;
  f[12]=(bf16_t)d.x; f[13]=(bf16_t)d.y; f[14]=(bf16_t)d.z; f[15]=(bf16_t)d.w;
  return f;
}

__device__ inline float rmax16(float v) {
#pragma unroll
  for (int m = 1; m < 16; m <<= 1) v = fmaxf(v, __shfl_xor(v, m, 32));
  return v;
}
__device__ inline float rsum16(float v) {
#pragma unroll
  for (int m = 1; m < 16; m <<= 1) v += __shfl_xor(v, m, 32);
  return v;
}

// ---------------------------------------------------------------------------
// Generic A/B fragment builder from a row-major fp32 matrix [rows][ld].
// One wave per (row-tile, k-chunk).  Fragment element e of lane L holds
// src[rt*16 + L%16][kc*32 + e + (e>=8?8:0) + (L>=16?8:0)].
// ---------------------------------------------------------------------------
__global__ void __launch_bounds__(32)
frag_build(const float* __restrict__ src, v16bf* __restrict__ dst,
           int kchunks, int ld) {
  int id   = blockIdx.x;
  int kc   = id % kchunks;
  int rt   = id / kchunks;
  int lane = threadIdx.x;
  int hi   = lane >> 4;
  const float* p = src + (long)(rt * 16 + (lane & 15)) * ld + kc * 32 + hi * 8;
  float4 r0 = *(const float4*)(p);
  float4 r1 = *(const float4*)(p + 4);
  float4 r2 = *(const float4*)(p + 16);
  float4 r3 = *(const float4*)(p + 20);
  dst[(long)id * 32 + lane] = pack16(r0, r1, r2, r3);
}

// ---------------------------------------------------------------------------
// Build Q (A-frag) and K (B-frag) fragments from qkv[12544][1152].
// qkv col layout: q at h*64, k at 384 + h*64, v at 768 + h*64.
// blockIdx = ((which*BH + bh)*QT + qt)*2 + c
// ---------------------------------------------------------------------------
__global__ void __launch_bounds__(32)
qk_frag(const float* __restrict__ qkv, v16bf* __restrict__ qf,
        v16bf* __restrict__ kf) {
  int id = blockIdx.x;
  int c  = id & 1;  id >>= 1;
  int qt = id % QT; id /= QT;
  int bh = id % BH;
  int which = id / BH;                 // 0 = q, 1 = k
  int b = bh / H_, h = bh % H_;
  int lane = threadIdx.x, hi = lane >> 4;
  int row = b * N_ + qt * 16 + (lane & 15);
  int colbase = which * 384 + h * 64 + c * 32 + hi * 8;
  const float* p = qkv + (long)row * 1152 + colbase;
  float4 r0 = *(const float4*)(p);
  float4 r1 = *(const float4*)(p + 4);
  float4 r2 = *(const float4*)(p + 16);
  float4 r3 = *(const float4*)(p + 20);
  v16bf fr = pack16(r0, r1, r2, r3);
  v16bf* dst = which ? kf : qf;
  dst[((long)(bh * QT + qt) * 2 + c) * 32 + lane] = fr;
}

// ---------------------------------------------------------------------------
// Build V B-fragments (transposed access) via LDS.
// For P@V: B matrix is [32 keys (K)] x [16 dcols (N)].
// blockIdx = bh*KGn + kg; one wave stages v[kg*32..+31][0..63] in LDS.
// ---------------------------------------------------------------------------
__global__ void __launch_bounds__(32)
v_frag(const float* __restrict__ qkv, v16bf* __restrict__ vf) {
  __shared__ float vsh[32][68];
  int kg = blockIdx.x % KGn;
  int bh = blockIdx.x / KGn;
  int b = bh / H_, h = bh % H_;
  int lane = threadIdx.x;
  const float* src = qkv + (long)(b * N_ + kg * 32 + lane) * 1152 + 768 + h * 64;
#pragma unroll
  for (int i = 0; i < 16; ++i) {
    float4 t = *(const float4*)(src + i * 4);
    vsh[lane][i * 4 + 0] = t.x; vsh[lane][i * 4 + 1] = t.y;
    vsh[lane][i * 4 + 2] = t.z; vsh[lane][i * 4 + 3] = t.w;
  }
  __syncthreads();
  int hi = lane >> 4, n = lane & 15;
#pragma unroll
  for (int dt = 0; dt < 4; ++dt) {
    v16bf fr;
#pragma unroll
    for (int e = 0; e < 16; ++e) {
      int k = e + ((e >= 8) ? 8 : 0) + hi * 8;
      fr[e] = (bf16_t)vsh[k][dt * 16 + n];
    }
    vf[((long)(bh * KGn + kg) * 4 + dt) * 32 + lane] = fr;
  }
}

// ---------------------------------------------------------------------------
// Tiled WMMA GEMM: out[M x (ngroups*64)] = A @ B^T (+ bias), fp32 out.
// One wave computes a 16x64 tile (4 accumulators), chaining WMMAs over K.
// ---------------------------------------------------------------------------
__global__ void __launch_bounds__(256)
gemm_bf16(const v16bf* __restrict__ A, const v16bf* __restrict__ Bf,
          const float* __restrict__ bias, float* __restrict__ out,
          int ngroups, int kchunks, int ldc) {
  int wid  = (blockIdx.x * 256 + threadIdx.x) >> 5;
  int lane = threadIdx.x & 31;
  int hi   = lane >> 4;
  int mt = wid / ngroups, ng = wid % ngroups;

  v8f c[4];
#pragma unroll
  for (int t = 0; t < 4; ++t) c[t] = (v8f){0, 0, 0, 0, 0, 0, 0, 0};

  const v16bf* ap = A + (long)mt * kchunks * 32 + lane;
  const v16bf* bp[4];
#pragma unroll
  for (int t = 0; t < 4; ++t)
    bp[t] = Bf + (long)(ng * 4 + t) * kchunks * 32 + lane;

  for (int kc = 0; kc < kchunks; ++kc) {
    v16bf a = ap[kc * 32];
#pragma unroll
    for (int t = 0; t < 4; ++t)
      c[t] = WMMA_BF16(a, bp[t][kc * 32], c[t]);
  }

  int col0 = ng * 64 + (lane & 15);
#pragma unroll
  for (int t = 0; t < 4; ++t) {
    int col = col0 + t * 16;
    float bv = bias ? bias[col] : 0.0f;
#pragma unroll
    for (int j = 0; j < 8; ++j) {
      int row = mt * 16 + j + 8 * hi;
      out[(long)row * ldc + col] = c[t][j] + bv;
    }
  }
}

// ---------------------------------------------------------------------------
// Flash attention with policy softmax. One wave owns a 16-query tile of one
// (b,h); streams 49 key-groups of 32.  Per group:
//   S = Q K^T (2x2 chained WMMAs), scale, raw-row-max (pre-mask, like ref),
//   online rescale, e = exp(S-m)*policy (diag always kept),
//   P = e + EPS/N  -> LDS transpose -> A-frag -> 4 WMMAs into O accumulator.
// Final: O / (l + EPS).
// ---------------------------------------------------------------------------
__global__ void __launch_bounds__(128)
attn_kernel(const v16bf* __restrict__ qf, const v16bf* __restrict__ kf,
            const v16bf* __restrict__ vf, const float* __restrict__ vis,
            float* __restrict__ O) {
  __shared__ float plds[4][16 * 36];
  int wid  = (blockIdx.x * 128 + threadIdx.x) >> 5;
  int w    = threadIdx.x >> 5;
  int lane = threadIdx.x & 31;
  int hi   = lane >> 4;
  int col  = lane & 15;
  int bh = wid / QT, qt = wid % QT;
  int b = bh / H_, h = bh % H_;
  int qbase = qt * 16;

  v16bf qa0 = qf[((long)(bh * QT + qt) * 2 + 0) * 32 + lane];
  v16bf qa1 = qf[((long)(bh * QT + qt) * 2 + 1) * 32 + lane];

  float m[8], l[8];
  v8f acc[4];
#pragma unroll
  for (int j = 0; j < 8; ++j) { m[j] = -3.0e38f; l[j] = 0.0f; }
#pragma unroll
  for (int t = 0; t < 4; ++t) acc[t] = (v8f){0, 0, 0, 0, 0, 0, 0, 0};

  float* myl = &plds[w][0];

  for (int kg = 0; kg < KGn; ++kg) {
    int kt0 = kg * 2, kt1 = kg * 2 + 1;
    v8f s0 = (v8f){0, 0, 0, 0, 0, 0, 0, 0};
    v8f s1 = (v8f){0, 0, 0, 0, 0, 0, 0, 0};
    s0 = WMMA_BF16(qa0, kf[((long)(bh * QT + kt0) * 2 + 0) * 32 + lane], s0);
    s0 = WMMA_BF16(qa1, kf[((long)(bh * QT + kt0) * 2 + 1) * 32 + lane], s0);
    s1 = WMMA_BF16(qa0, kf[((long)(bh * QT + kt1) * 2 + 0) * 32 + lane], s1);
    s1 = WMMA_BF16(qa1, kf[((long)(bh * QT + kt1) * 2 + 1) * 32 + lane], s1);

    int key0 = kg * 32 + col, key1 = key0 + 16;
    float pol0 = vis[b * N_ + key0];
    float pol1 = vis[b * N_ + key1];

    float e0[8], e1[8], fct[8];
#pragma unroll
    for (int j = 0; j < 8; ++j) {
      float a0 = s0[j] * SCALE, a1 = s1[j] * SCALE;
      float tmax = rmax16(fmaxf(a0, a1));       // raw max, pre-mask (as ref)
      float mn = fmaxf(m[j], tmax);
      fct[j] = __expf(m[j] - mn);
      int rq = qbase + j + 8 * hi;
      float ap0 = pol0 + (1.0f - pol0) * ((key0 == rq) ? 1.0f : 0.0f);
      float ap1 = pol1 + (1.0f - pol1) * ((key1 == rq) ? 1.0f : 0.0f);
      e0[j] = __expf(a0 - mn) * ap0;
      e1[j] = __expf(a1 - mn) * ap1;
      l[j] = l[j] * fct[j] + rsum16(e0[j] + e1[j]);
      m[j] = mn;
    }
#pragma unroll
    for (int t = 0; t < 4; ++t)
#pragma unroll
      for (int j = 0; j < 8; ++j) acc[t][j] *= fct[j];

    // C-layout -> LDS [16 rows][36 floats] (padded, conflict-free)
#pragma unroll
    for (int j = 0; j < 8; ++j) {
      int r = (j + 8 * hi) * 36;
      myl[r + col]      = e0[j] + EPSN;
      myl[r + 16 + col] = e1[j] + EPSN;
    }
    asm volatile("s_wait_dscnt 0x0" ::: "memory");

    // Reload as A-frag: lane = query row, elems = keys {k0..k0+7, k0+16..+23}
    int k0 = hi * 8;
    float4 p0 = *(const float4*)(&myl[col * 36 + k0]);
    float4 p1 = *(const float4*)(&myl[col * 36 + k0 + 4]);
    float4 p2 = *(const float4*)(&myl[col * 36 + k0 + 16]);
    float4 p3 = *(const float4*)(&myl[col * 36 + k0 + 20]);
    v16bf pa = pack16(p0, p1, p2, p3);

#pragma unroll
    for (int t = 0; t < 4; ++t)
      acc[t] = WMMA_BF16(pa, vf[((long)(bh * KGn + kg) * 4 + t) * 32 + lane], acc[t]);
  }

  float inv[8];
#pragma unroll
  for (int j = 0; j < 8; ++j) inv[j] = 1.0f / (l[j] + EPS);
#pragma unroll
  for (int t = 0; t < 4; ++t)
#pragma unroll
    for (int j = 0; j < 8; ++j) {
      int row = b * N_ + qbase + j + 8 * hi;
      O[(long)row * C_ + h * D_ + t * 16 + col] = acc[t][j] * inv[j];
    }
}

// ---------------------------------------------------------------------------
extern "C" void kernel_launch(void* const* d_in, const int* in_sizes, int n_in,
                              void* d_out, int out_size, void* d_ws, size_t ws_size,
                              hipStream_t stream) {
  const float* x      = (const float*)d_in[0];
  const float* vis    = (const float*)d_in[1];
  const float* w_qkv  = (const float*)d_in[2];
  const float* w_proj = (const float*)d_in[3];
  const float* b_proj = (const float*)d_in[4];

  char* ws = (char*)d_ws;
  v16bf* xf    = (v16bf*)(ws + OFF_XF);
  v16bf* wqkvf = (v16bf*)(ws + OFF_WQKV);
  v16bf* wprjf = (v16bf*)(ws + OFF_WPRJ);
  v16bf* qfr   = (v16bf*)(ws + OFF_QF);
  v16bf* kfr   = (v16bf*)(ws + OFF_KF);
  v16bf* vfr   = (v16bf*)(ws + OFF_VF);
  float* qkv   = (float*)(ws + OFF_QKV);
  float* Obuf  = (float*)(ws + OFF_O);
  v16bf* ofr   = (v16bf*)(ws + OFF_OF);

  // 1) fragment inputs/weights
  frag_build<<<784 * 12, 32, 0, stream>>>(x, xf, 12, 384);         // X A-frags
  frag_build<<<72 * 12, 32, 0, stream>>>(w_qkv, wqkvf, 12, 384);   // Wqkv B-frags
  frag_build<<<24 * 12, 32, 0, stream>>>(w_proj, wprjf, 12, 384);  // Wproj B-frags

  // 2) QKV projection: [12544x384] @ [384x1152] -> qkv fp32
  gemm_bf16<<<1764, 256, 0, stream>>>(xf, wqkvf, nullptr, qkv, 18, 12, 1152);

  // 3) per-head fragments for attention
  qk_frag<<<2 * BH * QT * 2, 32, 0, stream>>>(qkv, qfr, kfr);      // 18816 waves
  v_frag<<<BH * KGn, 32, 0, stream>>>(qkv, vfr);                   // 2352 waves

  // 4) flash attention -> O fp32 [12544x384] (reuses dead qkv region)
  attn_kernel<<<(BH * QT) / 4, 128, 0, stream>>>(qfr, kfr, vfr, vis, Obuf);

  // 5) output projection: O @ Wproj^T + b
  frag_build<<<784 * 12, 32, 0, stream>>>(Obuf, ofr, 12, 384);
  gemm_bf16<<<588, 256, 0, stream>>>(ofr, wprjf, b_proj, (float*)d_out, 6, 12, 384);
}
